// RWKV_Tmix_x060f_20830591385837
// MI455X (gfx1250) — matrix operand order repaired and checked by reading, not compiled, not run on hardware
//
#include <hip/hip_runtime.h>
#include <hip/hip_bf16.h>

typedef unsigned short u16;
typedef __attribute__((ext_vector_type(16))) __bf16 v16bf;
typedef __attribute__((ext_vector_type(8)))  __bf16 v8bf;
typedef __attribute__((ext_vector_type(8)))  float  v8f;

// ---------- helpers ----------
__device__ __forceinline__ u16 f2bf(float f) {
    union { float f; unsigned u; } x; x.f = f;
    unsigned r = x.u + 0x7fffu + ((x.u >> 16) & 1u);   // round-to-nearest-even
    return (u16)(r >> 16);
}

// ---------- constants ----------
#define CC   1024
#define TT   1024
#define BB   4
#define BT   4096        // B*T
#define HH   16
#define KK   64
#define DF   2048
#define NCAT 3072        // C + DF

// =====================================================================
// 0) weight pack: fp32 [K][N] -> bf16 Bp[k/32][n][k%32]
//    => per-lane WMMA B fragment (16 consecutive K at one column) is a
//       single aligned 32-byte global load.
// =====================================================================
__global__ __launch_bounds__(256) void pack_b_kernel(const float* __restrict__ src,
                                                     u16* __restrict__ dst, int K, int N) {
    int i = blockIdx.x * 256 + threadIdx.x;
    if (i >= K * N) return;
    int k = i / N, n = i - k * N;
    dst[((size_t)(k >> 5) * N + n) * 32 + (k & 31)] = f2bf(src[i]);
}

// =====================================================================
// 1) token prep: shift-mix + low-rank MLP -> xr/xk/xv (bf16, row-major)
// =====================================================================
__global__ __launch_bounds__(256) void prep_kernel(
    const float* __restrict__ x,
    const float* __restrict__ maa_x, const float* __restrict__ maa_k,
    const float* __restrict__ maa_v, const float* __restrict__ maa_r,
    const float* __restrict__ w1, const float* __restrict__ w2,
    u16* __restrict__ xr_bf, u16* __restrict__ xk_bf, u16* __restrict__ xv_bf)
{
    __shared__ float xs[CC], xxs[CC], xxxs[CC];
    __shared__ float h[96];
    int n = blockIdx.x;                 // token id in [0, BT)
    int t = n & (TT - 1);
    int tid = threadIdx.x;
    const float* xrow = x + (size_t)n * CC;

    for (int c = tid; c < CC; c += 256) {
        float xc = xrow[c];
        float xp = (t == 0) ? 0.f : xrow[c - CC];
        float xx = xp - xc;
        xs[c] = xc; xxs[c] = xx;
        xxxs[c] = fmaf(xx, maa_x[c], xc);
    }
    __syncthreads();
    if (tid < 96) {
        float acc = 0.f;
        for (int c = 0; c < CC; ++c) acc = fmaf(xxxs[c], w1[c * 96 + tid], acc);
        h[tid] = tanhf(acc);
    }
    __syncthreads();
    for (int c = tid; c < CC; c += 256) {
        float mk = 0.f, mv = 0.f, mr = 0.f;
#pragma unroll 8
        for (int d = 0; d < 32; ++d) {
            mk = fmaf(h[d],      w2[(0 * 32 + d) * CC + c], mk);
            mv = fmaf(h[32 + d], w2[(1 * 32 + d) * CC + c], mv);
            mr = fmaf(h[64 + d], w2[(2 * 32 + d) * CC + c], mr);
        }
        float xc = xs[c], xx = xxs[c];
        size_t o = (size_t)n * CC + c;
        xk_bf[o] = f2bf(fmaf(xx, maa_k[c] + mk, xc));
        xv_bf[o] = f2bf(fmaf(xx, maa_v[c] + mv, xc));
        xr_bf[o] = f2bf(fmaf(xx, maa_r[c] + mr, xc));
    }
}

// ---------- fragment helpers ----------
__device__ __forceinline__ v16bf load_a_frag(const u16* __restrict__ arow, int khiA) {
    v8bf alo = *reinterpret_cast<const v8bf*>(arow + khiA);        // K = khiA..khiA+7
    v8bf ahi = *reinterpret_cast<const v8bf*>(arow + 8 + khiA);    // K = 8+khiA..+7
    return __builtin_shufflevector(alo, ahi, 0,1,2,3,4,5,6,7,8,9,10,11,12,13,14,15);
}

// =====================================================================
// 2) WMMA bf16 GEMM: C[M,N] = A[M,K] * Bp; block tile 64x128,
//    8 waves (4 row x 2 col groups), each wave 16x64 = 4 WMMA / k-step.
//    A via LDS (2x ds_load_b128 per frag); B direct from packed global
//    (one aligned 32B load per fragment).
//    EPI 0: plain fp32 store;  EPI 1: e = exp(-exp(acc)).
// =====================================================================
template<int EPI>
__global__ __launch_bounds__(256) void gemm_bf16(
    const u16* __restrict__ A, const u16* __restrict__ Bp,
    float* __restrict__ Cf, int K, int N, int ldc)
{
    __shared__ u16 As[64][32];

    const int t     = threadIdx.x;
    const int w     = t >> 5;
    const int lane  = t & 31;
    const int mTile = blockIdx.x * 64;
    const int nTile = blockIdx.y * 128;

    const int lm    = lane & 15;
    const int khiA  = (lane >> 4) * 16;
    const int khiB  = (lane >> 4) * 16;
    const int colB  = nTile + (w >> 2) * 64 + lm;   // tiles at +0,+16,+32,+48
    const int arow  = t >> 2, aseg = t & 3;

    v8f acc[4];
#pragma unroll
    for (int j = 0; j < 4; ++j) acc[j] = (v8f){0.f,0.f,0.f,0.f,0.f,0.f,0.f,0.f};

    for (int kb = 0; kb < K; kb += 32) {
        *reinterpret_cast<uint4*>(&As[arow][aseg * 8]) =
            *(reinterpret_cast<const uint4*>(A + (size_t)(mTile + arow) * K + kb) + aseg);
        if (kb + 32 < K)
            __builtin_prefetch(A + (size_t)(mTile + arow) * K + (kb + 32), 0, 3);
        __syncthreads();

        v16bf a = load_a_frag(&As[(w & 3) * 16 + lm][0], khiA);

        const u16* bcol = Bp + ((size_t)(kb >> 5) * N + colB) * 32 + khiB;
        v16bf b0 = *reinterpret_cast<const v16bf*>(bcol);
        v16bf b1 = *reinterpret_cast<const v16bf*>(bcol + 16 * 32);
        v16bf b2 = *reinterpret_cast<const v16bf*>(bcol + 32 * 32);
        v16bf b3 = *reinterpret_cast<const v16bf*>(bcol + 48 * 32);

        acc[0] = __builtin_amdgcn_wmma_f32_16x16x32_bf16(false, a, false, b0, (short)0, acc[0], false, false);
        acc[1] = __builtin_amdgcn_wmma_f32_16x16x32_bf16(false, a, false, b1, (short)0, acc[1], false, false);
        acc[2] = __builtin_amdgcn_wmma_f32_16x16x32_bf16(false, a, false, b2, (short)0, acc[2], false, false);
        acc[3] = __builtin_amdgcn_wmma_f32_16x16x32_bf16(false, a, false, b3, (short)0, acc[3], false, false);
        __syncthreads();
    }

    const int rowBase = mTile + (w & 3) * 16 + ((lane >> 4) * 8);
#pragma unroll
    for (int j = 0; j < 4; ++j) {
        int col = colB + 16 * j;
#pragma unroll
        for (int i = 0; i < 8; ++i) {
            float v0 = acc[j][i];
            if (EPI == 1) v0 = expf(-expf(v0));
            Cf[(size_t)(rowBase + i) * ldc + col] = v0;
        }
    }
}

// =====================================================================
// 3) vg GEMM: columns j and j+3072 computed together, SiLU-gated.
//    Block tile 64x64 (value cols) + paired gate cols; packed B (N=6144).
//    j <  C -> v (fp32);  j >= C -> ffn (bf16 into concat[:, C:]).
// =====================================================================
__global__ __launch_bounds__(256) void gemm_vg(
    const u16* __restrict__ A, const u16* __restrict__ Bp,
    float* __restrict__ vbuf, u16* __restrict__ concat, int K, int Npk)
{
    __shared__ u16 As[64][32];

    const int t     = threadIdx.x;
    const int w     = t >> 5;
    const int lane  = t & 31;
    const int mTile = blockIdx.x * 64;
    const int nTile = blockIdx.y * 64;          // value-column tile in [0, 3072)

    const int lm    = lane & 15;
    const int khiA  = (lane >> 4) * 16;
    const int khiB  = (lane >> 4) * 16;
    const int colB  = nTile + (w >> 2) * 32 + lm;   // value tiles at +0,+16
    const int arow  = t >> 2, aseg = t & 3;

    v8f aV0 = {0.f,0.f,0.f,0.f,0.f,0.f,0.f,0.f};
    v8f aV1 = {0.f,0.f,0.f,0.f,0.f,0.f,0.f,0.f};
    v8f aG0 = {0.f,0.f,0.f,0.f,0.f,0.f,0.f,0.f};
    v8f aG1 = {0.f,0.f,0.f,0.f,0.f,0.f,0.f,0.f};

    for (int kb = 0; kb < K; kb += 32) {
        *reinterpret_cast<uint4*>(&As[arow][aseg * 8]) =
            *(reinterpret_cast<const uint4*>(A + (size_t)(mTile + arow) * K + kb) + aseg);
        __syncthreads();

        v16bf a = load_a_frag(&As[(w & 3) * 16 + lm][0], khiA);

        const u16* bcol = Bp + ((size_t)(kb >> 5) * Npk + colB) * 32 + khiB;
        v16bf b0 = *reinterpret_cast<const v16bf*>(bcol);
        v16bf b1 = *reinterpret_cast<const v16bf*>(bcol + 16 * 32);
        v16bf g0 = *reinterpret_cast<const v16bf*>(bcol + (size_t)NCAT * 32);
        v16bf g1 = *reinterpret_cast<const v16bf*>(bcol + (size_t)(NCAT + 16) * 32);

        aV0 = __builtin_amdgcn_wmma_f32_16x16x32_bf16(false, a, false, b0, (short)0, aV0, false, false);
        aV1 = __builtin_amdgcn_wmma_f32_16x16x32_bf16(false, a, false, b1, (short)0, aV1, false, false);
        aG0 = __builtin_amdgcn_wmma_f32_16x16x32_bf16(false, a, false, g0, (short)0, aG0, false, false);
        aG1 = __builtin_amdgcn_wmma_f32_16x16x32_bf16(false, a, false, g1, (short)0, aG1, false, false);
        __syncthreads();
    }

    const int rowBase = mTile + (w & 3) * 16 + ((lane >> 4) * 8);
#pragma unroll
    for (int i = 0; i < 8; ++i) {
        int row = rowBase + i;
        {   // tile 0
            float g = aG0[i];
            float val = aV0[i] * (g / (1.f + expf(-g)));
            if (colB < CC) vbuf[(size_t)row * CC + colB] = val;
            else           concat[(size_t)row * NCAT + colB] = f2bf(val);
        }
        {   // tile 1
            int c1 = colB + 16;
            float g = aG1[i];
            float val = aV1[i] * (g / (1.f + expf(-g)));
            if (c1 < CC) vbuf[(size_t)row * CC + c1] = val;
            else         concat[(size_t)row * NCAT + c1] = f2bf(val);
        }
    }
}

// =====================================================================
// 4) WKV6 scan: one block per (b,h); thread (chunk,vi) holds S[16] regs.
// =====================================================================
__global__ __launch_bounds__(256) void wkv_scan(
    const float* __restrict__ r, const float* __restrict__ e,
    const float* __restrict__ v, const float* __restrict__ u,
    float* __restrict__ o)
{
    __shared__ float r_sh[KK], e_sh[KK], v_sh[KK], u_sh[KK], part[256];
    const int bh = blockIdx.x;
    const int b = bh >> 4, h = bh & 15;
    const int tid   = threadIdx.x;
    const int chunk = tid >> 6;        // 0..3  -> k in [chunk*16, chunk*16+16)
    const int vi    = tid & 63;
    if (tid < KK) u_sh[tid] = u[h * KK + tid];

    float S[16];
#pragma unroll
    for (int i = 0; i < 16; ++i) S[i] = 0.f;

    const size_t base = ((size_t)b * TT) * CC + (size_t)h * KK;
    for (int t = 0; t < TT; ++t) {
        size_t idx = base + (size_t)t * CC;
        if (tid < 64)       r_sh[tid]        = r[idx + tid];
        else if (tid < 128) e_sh[tid - 64]   = e[idx + (tid - 64)];
        else if (tid < 192) v_sh[tid - 128]  = v[idx + (tid - 128)];
        __syncthreads();

        const float vv = v_sh[vi];
        const int kb = chunk << 4;
        float acc = 0.f;
#pragma unroll
        for (int i = 0; i < 16; ++i) {
            float ev = e_sh[kb + i];
            float kv = (1.f - ev) * vv;
            acc = fmaf(r_sh[kb + i], fmaf(u_sh[kb + i], kv, S[i]), acc);
            S[i] = fmaf(S[i], ev, kv);
        }
        part[tid] = acc;
        __syncthreads();
        if (tid < 64)
            o[idx + tid] = part[tid] + part[64 + tid] + part[128 + tid] + part[192 + tid];
    }
}

// =====================================================================
// 5) LayerNorm over o -> bf16 into concat[:, :C]
// =====================================================================
__global__ __launch_bounds__(256) void ln_kernel(
    const float* __restrict__ o_raw, const float* __restrict__ g,
    const float* __restrict__ bta, u16* __restrict__ concat)
{
    __shared__ float red[256];
    const int n = blockIdx.x, tid = threadIdx.x;
    const float* row = o_raw + (size_t)n * CC;
    float s = 0.f;
    for (int c = tid; c < CC; c += 256) s += row[c];
    red[tid] = s; __syncthreads();
    for (int off = 128; off > 0; off >>= 1) { if (tid < off) red[tid] += red[tid + off]; __syncthreads(); }
    const float mu = red[0] * (1.f / CC);
    __syncthreads();
    float var = 0.f;
    for (int c = tid; c < CC; c += 256) { float d = row[c] - mu; var += d * d; }
    red[tid] = var; __syncthreads();
    for (int off = 128; off > 0; off >>= 1) { if (tid < off) red[tid] += red[tid + off]; __syncthreads(); }
    const float rstd = rsqrtf(red[0] * (1.f / CC) + 1e-5f);
    for (int c = tid; c < CC; c += 256)
        concat[(size_t)n * NCAT + c] = f2bf((row[c] - mu) * rstd * g[c] + bta[c]);
}

// =====================================================================
// launch
// =====================================================================
extern "C" void kernel_launch(void* const* d_in, const int* in_sizes, int n_in,
                              void* d_out, int out_size, void* d_ws, size_t ws_size,
                              hipStream_t stream) {
    (void)in_sizes; (void)n_in; (void)out_size; (void)ws_size;
    const float* x     = (const float*)d_in[0];
    const float* maa_x = (const float*)d_in[1];
    const float* maa_k = (const float*)d_in[2];
    const float* maa_v = (const float*)d_in[3];
    const float* maa_r = (const float*)d_in[4];
    const float* w1    = (const float*)d_in[5];
    const float* w2    = (const float*)d_in[6];
    const float* u     = (const float*)d_in[7];
    const float* Wr    = (const float*)d_in[8];
    const float* Wk    = (const float*)d_in[9];
    const float* Wvg   = (const float*)d_in[10];
    const float* Wo    = (const float*)d_in[11];
    const float* ln_g  = (const float*)d_in[12];
    const float* ln_b  = (const float*)d_in[13];
    float* out = (float*)d_out;

    char* p = (char*)d_ws;
    auto take = [&](size_t bytes) { char* q = p; p += (bytes + 255) & ~(size_t)255; return q; };
    u16*   xr_bf  = (u16*)  take((size_t)BT * CC * 2);
    u16*   xk_bf  = (u16*)  take((size_t)BT * CC * 2);
    u16*   xv_bf  = (u16*)  take((size_t)BT * CC * 2);
    u16*   Wr_pk  = (u16*)  take((size_t)CC * CC * 2);
    u16*   Wk_pk  = (u16*)  take((size_t)CC * CC * 2);
    u16*   Wvg_pk = (u16*)  take((size_t)CC * 2 * NCAT * 2);
    u16*   Wo_pk  = (u16*)  take((size_t)NCAT * CC * 2);
    float* r_f    = (float*)take((size_t)BT * CC * 4);
    float* e_f    = (float*)take((size_t)BT * CC * 4);
    float* v_f    = (float*)take((size_t)BT * CC * 4);
    float* o_raw  = (float*)take((size_t)BT * CC * 4);
    u16*   concat = (u16*)  take((size_t)BT * NCAT * 2);

    // 0) weight packing (WMMA-fragment-friendly layout)
    {
        int n1 = CC * CC;
        pack_b_kernel<<<(n1 + 255) / 256, 256, 0, stream>>>(Wr, Wr_pk, CC, CC);
        pack_b_kernel<<<(n1 + 255) / 256, 256, 0, stream>>>(Wk, Wk_pk, CC, CC);
        int n2 = CC * 2 * NCAT;
        pack_b_kernel<<<(n2 + 255) / 256, 256, 0, stream>>>(Wvg, Wvg_pk, CC, 2 * NCAT);
        int n3 = NCAT * CC;
        pack_b_kernel<<<(n3 + 255) / 256, 256, 0, stream>>>(Wo, Wo_pk, NCAT, CC);
    }
    // 1) token prep
    prep_kernel<<<BT, 256, 0, stream>>>(x, maa_x, maa_k, maa_v, maa_r, w1, w2,
                                        xr_bf, xk_bf, xv_bf);
    // 2) r = xr @ Wr ; e = exp(-exp(xk @ Wk))
    gemm_bf16<0><<<dim3(BT / 64, CC / 128), 256, 0, stream>>>(xr_bf, Wr_pk, r_f, CC, CC, CC);
    gemm_bf16<1><<<dim3(BT / 64, CC / 128), 256, 0, stream>>>(xk_bf, Wk_pk, e_f, CC, CC, CC);
    // 3) vg = xv @ Wvg, gated -> v (fp32) + ffn (bf16 into concat)
    gemm_vg<<<dim3(BT / 64, NCAT / 64), 256, 0, stream>>>(xv_bf, Wvg_pk, v_f, concat, CC, 2 * NCAT);
    // 4) WKV scan
    wkv_scan<<<BB * HH, 256, 0, stream>>>(r_f, e_f, v_f, u, o_raw);
    // 5) LayerNorm -> concat[:, :C]
    ln_kernel<<<BT, 256, 0, stream>>>(o_raw, ln_g, ln_b, concat);
    // 6) out = concat @ Wo
    gemm_bf16<0><<<dim3(BT / 64, CC / 128), 256, 0, stream>>>(concat, Wo_pk, out, NCAT, CC, CC);
}
